// FPModule_85839216377950
// MI455X (gfx1250) — compile-verified, matrix operand order checked by reading, not run on hardware
//
#include <hip/hip_runtime.h>
#include <hip/hip_bf16.h>

typedef __attribute__((ext_vector_type(16))) _Float16 v16h;
typedef __attribute__((ext_vector_type(8)))  float    v8f;

#define NF      65536      // fine points (8 * 8192)
#define NC_PER  2048       // coarse points per batch
#define NF_PER  8192       // fine points per batch
#define CIN     384        // concat feature dim
#define CH1     512        // hidden dim
#define CH2     256        // output channels
#define GN_EPS  1e-5f

// ---------------------------------------------------------------------------
// 1) Brute-force kNN (k=3) with LDS-tiled coarse positions, inverse-distance
//    weights normalized in-kernel.
// ---------------------------------------------------------------------------
__global__ __launch_bounds__(256) void knn_kernel(const float* __restrict__ pos,
                                                  const float* __restrict__ pos_skip,
                                                  int* __restrict__ idx_out,
                                                  float* __restrict__ w_out) {
    __shared__ float sx[256], sy[256], sz[256];
    const int f = blockIdx.x * 256 + threadIdx.x;
    const int b = f >> 13;                 // f / 8192
    const int cbase = b << 11;             // b * 2048

    const float px = pos_skip[(size_t)f * 3 + 0];
    const float py = pos_skip[(size_t)f * 3 + 1];
    const float pz = pos_skip[(size_t)f * 3 + 2];

    float d0 = 3.4e38f, d1 = 3.4e38f, d2 = 3.4e38f;
    int   i0 = cbase, i1 = cbase, i2 = cbase;

    for (int tile = 0; tile < NC_PER; tile += 256) {
        const int c = cbase + tile + threadIdx.x;
        sx[threadIdx.x] = pos[(size_t)c * 3 + 0];
        sy[threadIdx.x] = pos[(size_t)c * 3 + 1];
        sz[threadIdx.x] = pos[(size_t)c * 3 + 2];
        __syncthreads();
        #pragma unroll 8
        for (int j = 0; j < 256; ++j) {
            const float dx = px - sx[j];
            const float dy = py - sy[j];
            const float dz = pz - sz[j];
            const float d = dx * dx + dy * dy + dz * dz;
            const int ci = cbase + tile + j;
            if (d < d2) {
                if (d < d0)      { d2 = d1; i2 = i1; d1 = d0; i1 = i0; d0 = d; i0 = ci; }
                else if (d < d1) { d2 = d1; i2 = i1; d1 = d;  i1 = ci; }
                else             { d2 = d;  i2 = ci; }
            }
        }
        __syncthreads();
    }

    const float w0 = 1.0f / fmaxf(d0, 1e-16f);
    const float w1 = 1.0f / fmaxf(d1, 1e-16f);
    const float w2 = 1.0f / fmaxf(d2, 1e-16f);
    const float inv_s = 1.0f / (w0 + w1 + w2);
    idx_out[(size_t)f * 3 + 0] = i0;
    idx_out[(size_t)f * 3 + 1] = i1;
    idx_out[(size_t)f * 3 + 2] = i2;
    w_out[(size_t)f * 3 + 0] = w0 * inv_s;
    w_out[(size_t)f * 3 + 1] = w1 * inv_s;
    w_out[(size_t)f * 3 + 2] = w2 * inv_s;
}

// ---------------------------------------------------------------------------
// 2) Build A = [interp(x) | x_skip] as f16, row-major [NF, 384].
// ---------------------------------------------------------------------------
__global__ __launch_bounds__(128) void build_a_kernel(const float* __restrict__ x,
                                                      const float* __restrict__ x_skip,
                                                      const int* __restrict__ idx,
                                                      const float* __restrict__ w,
                                                      _Float16* __restrict__ A) {
    const int f = blockIdx.x;
    const int t = threadIdx.x;
    const int i0 = idx[(size_t)f * 3 + 0];
    const int i1 = idx[(size_t)f * 3 + 1];
    const int i2 = idx[(size_t)f * 3 + 2];
    const float w0 = w[(size_t)f * 3 + 0];
    const float w1 = w[(size_t)f * 3 + 1];
    const float w2 = w[(size_t)f * 3 + 2];
    const float* x0 = x + (size_t)i0 * 256;
    const float* x1 = x + (size_t)i1 * 256;
    const float* x2 = x + (size_t)i2 * 256;
    _Float16* row = A + (size_t)f * CIN;

    #pragma unroll
    for (int p = 0; p < 2; ++p) {
        const int c = t + p * 128;
        row[c] = (_Float16)(w0 * x0[c] + w1 * x1[c] + w2 * x2[c]);
    }
    row[256 + t] = (_Float16)x_skip[(size_t)f * 128 + t];
}

// ---------------------------------------------------------------------------
// 3) f32 -> f16 weight conversion
// ---------------------------------------------------------------------------
__global__ __launch_bounds__(256) void cvt_f16_kernel(const float* __restrict__ src,
                                                      _Float16* __restrict__ dst, int n) {
    const int i = blockIdx.x * 256 + threadIdx.x;
    if (i < n) dst[i] = (_Float16)src[i];
}

// ---------------------------------------------------------------------------
// WMMA fragment load: 16-bit A 16x32 layout (CDNA5 ISA 7.12.2).
// Per lane: m = lane%16 (in rowptr), half = lane/16; VGPR v holds K pair
//   k(v) = 16*(v>>2) + 8*half + 2*(v&3)   -> two contiguous 16B runs
//   => compiler coalesces into 2x global_load_b128 (verified in disasm).
// B uses the mirrored layout with n = lane%16 and W stored row-major (n,k).
// ---------------------------------------------------------------------------
__device__ __forceinline__ v16h load_frag16(const _Float16* rowptr, int half) {
    union { v16h h; unsigned u[8]; } frag;
    #pragma unroll
    for (int v = 0; v < 8; ++v) {
        const int k = 16 * (v >> 2) + 8 * half + 2 * (v & 3);
        frag.u[v] = *(const unsigned*)(rowptr + k);
    }
    return frag.h;
}

// ---------------------------------------------------------------------------
// 4/5) Tiled WMMA GEMM:  C[m,n] = relu( sum_k A[m,k] * W[n,k] + bias[n] )
//      Block = 256 threads = 8 waves; each wave computes a 32(M) x 64(N)
//      strip (2 M-tiles x 4 N-tiles = 8 f32 accumulators), K-step 32.
//      B fragments are loaded just-in-time inside the MAC loop: this keeps
//      register pressure below the spill/un-unroll cliff (round-3 regression)
//      while WMMA execution in the XDL pipe overlaps the next fragment's
//      load latency. Verified codegen: x2-unrolled K-loop, 8 wmma + 12 b128
//      per step, no hazard v_nops.
//      F16OUT: store f16 to workspace (feeds GEMM2); else f32 to output.
// ---------------------------------------------------------------------------
template <bool F16OUT>
__global__ __launch_bounds__(256) void gemm_relu_kernel(const _Float16* __restrict__ A,
                                                        const _Float16* __restrict__ Wh,
                                                        const float* __restrict__ bias,
                                                        void* __restrict__ Cout,
                                                        int Kdim, int N) {
    const int lane = threadIdx.x & 31;
    const int wave = threadIdx.x >> 5;
    const int half = lane >> 4;
    const int l16  = lane & 15;

    const int mrow0 = blockIdx.x * 256 + wave * 32 + l16;  // M-tile 0 load row
    const int ncol0 = blockIdx.y * 64;

    v8f acc[2][4] = {};
    const _Float16* arow0 = A + (size_t)mrow0 * Kdim;
    const _Float16* arow1 = arow0 + (size_t)16 * Kdim;     // M-tile 1

    for (int k0 = 0; k0 < Kdim; k0 += 32) {
        const v16h af0 = load_frag16(arow0 + k0, half);
        const v16h af1 = load_frag16(arow1 + k0, half);
        #pragma unroll
        for (int i = 0; i < 4; ++i) {
            const _Float16* brow = Wh + (size_t)(ncol0 + i * 16 + l16) * Kdim + k0;
            const v16h bf = load_frag16(brow, half);
            acc[0][i] = __builtin_amdgcn_wmma_f32_16x16x32_f16(
                false, af0, false, bf, (short)0, acc[0][i], false, false);
            acc[1][i] = __builtin_amdgcn_wmma_f32_16x16x32_f16(
                false, af1, false, bf, (short)0, acc[1][i], false, false);
        }
    }

    // C/D layout: VGPR r -> M = 8*half + r, N = lane%16 (per 16x16 tile)
    const int mbase = blockIdx.x * 256 + wave * 32 + 8 * half;
    #pragma unroll
    for (int t = 0; t < 2; ++t) {
        #pragma unroll
        for (int i = 0; i < 4; ++i) {
            const int n = ncol0 + i * 16 + l16;
            const float bv = bias[n];
            #pragma unroll
            for (int r = 0; r < 8; ++r) {
                float v = acc[t][i][r] + bv;
                v = v > 0.0f ? v : 0.0f;
                const size_t m = (size_t)(mbase + t * 16 + r);
                if (F16OUT) {
                    ((_Float16*)Cout)[m * N + n] = (_Float16)v;
                } else {
                    ((float*)Cout)[m * N + n] = v;
                }
            }
        }
    }
}

// ---------------------------------------------------------------------------
// 6) GroupNorm(32 groups, 256 ch): one wave per row; lane g owns the 8
//    contiguous channels of group g. In-place on the f32 output.
// ---------------------------------------------------------------------------
__global__ __launch_bounds__(256) void gn_kernel(float* __restrict__ h,
                                                 const float* __restrict__ gamma,
                                                 const float* __restrict__ beta) {
    const int lane = threadIdx.x & 31;
    const int wave = threadIdx.x >> 5;
    const int row  = blockIdx.x * 8 + wave;

    float* p = h + (size_t)row * CH2 + lane * 8;
    float v[8];
    *(float4*)(v)     = *(const float4*)(p);
    *(float4*)(v + 4) = *(const float4*)(p + 4);

    float mu = 0.0f;
    #pragma unroll
    for (int j = 0; j < 8; ++j) mu += v[j];
    mu *= 0.125f;
    float var = 0.0f;
    #pragma unroll
    for (int j = 0; j < 8; ++j) { const float d = v[j] - mu; var += d * d; }
    var *= 0.125f;
    const float inv = rsqrtf(var + GN_EPS);

    const float* g  = gamma + lane * 8;
    const float* be = beta  + lane * 8;
    #pragma unroll
    for (int j = 0; j < 8; ++j) v[j] = (v[j] - mu) * inv * g[j] + be[j];

    *(float4*)(p)     = *(const float4*)(v);
    *(float4*)(p + 4) = *(const float4*)(v + 4);
}

// ---------------------------------------------------------------------------
// 7) Pass-through tail outputs: pos_skip, reflectance_skip, batch_skip
// ---------------------------------------------------------------------------
__global__ __launch_bounds__(256) void tail_kernel(const float* __restrict__ pos_skip,
                                                   const float* __restrict__ refl,
                                                   const long long* __restrict__ batch,
                                                   float* __restrict__ out) {
    const int i = blockIdx.x * 256 + threadIdx.x;
    const int NPOS = NF * 3;
    if (i < NPOS) out[i] = pos_skip[i];
    if (i < NF) {
        out[NPOS + i] = refl[i];
        out[NPOS + NF + i] = (float)batch[i];
    }
}

// ---------------------------------------------------------------------------
extern "C" void kernel_launch(void* const* d_in, const int* in_sizes, int n_in,
                              void* d_out, int out_size, void* d_ws, size_t ws_size,
                              hipStream_t stream) {
    const float*     x       = (const float*)d_in[0];       // [16384, 256]
    const float*     pos     = (const float*)d_in[1];       // [16384, 3]
    const long long* batch_s = (const long long*)d_in[7];   // [65536]
    const float*     x_skip  = (const float*)d_in[4];       // [65536, 128]
    const float*     pos_sk  = (const float*)d_in[5];       // [65536, 3]
    const float*     refl_sk = (const float*)d_in[6];       // [65536]
    const float*     W1      = (const float*)d_in[8];       // [512, 384]
    const float*     b1      = (const float*)d_in[9];       // [512]
    const float*     W2      = (const float*)d_in[10];      // [256, 512]
    const float*     b2      = (const float*)d_in[11];      // [256]
    const float*     gamma2  = (const float*)d_in[12];      // [256]
    const float*     beta2   = (const float*)d_in[13];      // [256]

    // Workspace carve-up (256B aligned)
    size_t off = 0;
    auto carve = [&](size_t bytes) {
        size_t o = off;
        off = (off + bytes + 255) & ~(size_t)255;
        return o;
    };
    char* ws = (char*)d_ws;
    _Float16* A_h16  = (_Float16*)(ws + carve((size_t)NF * CIN * sizeof(_Float16)));
    _Float16* C1_h16 = (_Float16*)(ws + carve((size_t)NF * CH1 * sizeof(_Float16)));
    _Float16* W1_h16 = (_Float16*)(ws + carve((size_t)CH1 * CIN * sizeof(_Float16)));
    _Float16* W2_h16 = (_Float16*)(ws + carve((size_t)CH2 * CH1 * sizeof(_Float16)));
    int*      knn_i  = (int*)     (ws + carve((size_t)NF * 3 * sizeof(int)));
    float*    knn_w  = (float*)   (ws + carve((size_t)NF * 3 * sizeof(float)));
    (void)ws_size; (void)n_in; (void)in_sizes; (void)out_size;

    float* h_out    = (float*)d_out;                 // [65536, 256]
    float* tail_out = h_out + (size_t)NF * CH2;      // pos(3) + refl(1) + batch(1)

    // 1) kNN
    knn_kernel<<<NF / 256, 256, 0, stream>>>(pos, pos_sk, knn_i, knn_w);

    // 3) weight conversion (same stream, runs after knn)
    cvt_f16_kernel<<<(CH1 * CIN + 255) / 256, 256, 0, stream>>>(W1, W1_h16, CH1 * CIN);
    cvt_f16_kernel<<<(CH2 * CH1 + 255) / 256, 256, 0, stream>>>(W2, W2_h16, CH2 * CH1);

    // 2) interpolate + concat -> A (f16)
    build_a_kernel<<<NF, 128, 0, stream>>>(x, x_skip, knn_i, knn_w, A_h16);

    // 4) GEMM1: [65536,384] x [384,512]^T + b1, ReLU -> f16
    {
        dim3 grid(NF / 256, CH1 / 64);
        gemm_relu_kernel<true><<<grid, 256, 0, stream>>>(A_h16, W1_h16, b1,
                                                         (void*)C1_h16, CIN, CH1);
    }

    // 5) GEMM2: [65536,512] x [512,256]^T + b2, ReLU -> f32 into d_out
    {
        dim3 grid(NF / 256, CH2 / 64);
        gemm_relu_kernel<false><<<grid, 256, 0, stream>>>(C1_h16, W2_h16, b2,
                                                          (void*)h_out, CH1, CH2);
    }

    // 6) GroupNorm in-place
    gn_kernel<<<NF / 8, 256, 0, stream>>>(h_out, gamma2, beta2);

    // 7) tail copies
    tail_kernel<<<(NF * 3 + 255) / 256, 256, 0, stream>>>(pos_sk, refl_sk, batch_s, tail_out);
}
